// UpsampleMHA_32341103739098
// MI455X (gfx1250) — matrix operand order repaired and checked by reading, not compiled, not run on hardware
//
#include <hip/hip_runtime.h>

// ---------------------------------------------------------------------------
// UpsampleMHA on MI455X (gfx1250): f16 WMMA pipeline, wave32.
//   B=2, QN=2048, NDS=1024, H=16, Dh=64, D=1024
// v3: - kv loop split into (<=2) masked boundary chunks + mask-free steady
//       state (band mask j >= i-2 only touches the first 64 kv positions of
//       each 32-aligned scan start).
//     - softmax in exp2 domain: scale = 0.125*log2(e), exp2f -> v_exp_f32,
//       one fewer v_mul per exponential.
//     - GEMMs: LDS-free, pre-transposed weights, 32x64 tile/wave (asm-clean).
//     - Attention: S^T = K*Q^T so score C-layout == P*V A-layout; P stays in
//       registers, zero LDS in the whole attention kernel.
// ---------------------------------------------------------------------------

typedef __attribute__((ext_vector_type(16))) _Float16 v16h;
typedef __attribute__((ext_vector_type(8)))  _Float16 v8h;
typedef __attribute__((ext_vector_type(4)))  _Float16 v4h;
typedef __attribute__((ext_vector_type(8)))  float    v8f;

#define QN   2048
#define NDS  1024
#define NH   16
#define DH   64
#define DM   1024   // D_IN == INNER == D_OUT

static __device__ inline v8f zero8() {
  v8f z;
#pragma unroll
  for (int i = 0; i < 8; ++i) z[i] = 0.0f;
  return z;
}

static __device__ inline v16h ld16(const _Float16* p0, const _Float16* p1) {
  v8h lo = *(const v8h*)p0;
  v8h hi = *(const v8h*)p1;
  v16h r;
#pragma unroll
  for (int t = 0; t < 8; ++t) { r[t] = lo[t]; r[t + 8] = hi[t]; }
  return r;
}

static __device__ inline v8f wmma16(v16h a, v16h b, v8f c) {
  // (neg_a, A, neg_b, B, c_mod, C, reuse_a, reuse_b)
  return __builtin_amdgcn_wmma_f32_16x16x32_f16(false, a, false, b, (short)0, c,
                                                false, false);
}

// -------------------------------- f32 -> f16 -------------------------------
__global__ void __launch_bounds__(256)
cvt_f32_f16(const float* __restrict__ src, _Float16* __restrict__ dst, int n) {
  int i = (blockIdx.x * 256 + threadIdx.x) * 4;
  if (i < n) {
    float4 v = *(const float4*)(src + i);
    v4h o;
    o[0] = (_Float16)v.x; o[1] = (_Float16)v.y;
    o[2] = (_Float16)v.z; o[3] = (_Float16)v.w;
    *(v4h*)(dst + i) = o;
  }
}

// ---------------- fused convert + transpose: Wt[n][k] <- W[k][n] -----------
__global__ void __launch_bounds__(256)
cvt_transpose_w(const float* __restrict__ W, _Float16* __restrict__ Wt) {
  int idx = blockIdx.x * 256 + threadIdx.x;   // 1M elements
  int k = idx & (DM - 1);
  int n = idx >> 10;
  Wt[idx] = (_Float16)W[(size_t)k * DM + n];
}

// ------------------------- Vt[b][h][d][j] from V[b][j][h*64+d] -------------
__global__ void __launch_bounds__(256)
transpose_v(const _Float16* __restrict__ V, _Float16* __restrict__ Vt) {
  int idx = blockIdx.x * 256 + threadIdx.x;   // B*H*64*NDS = 2M
  int j = idx & (NDS - 1);
  int d = (idx >> 10) & 63;
  int h = (idx >> 16) & 15;
  int b = idx >> 20;
  Vt[idx] = V[((size_t)(b * NDS + j) << 10) + h * 64 + d];
}

// ------------------------------- WMMA GEMM ---------------------------------
// C[M,1024] = A[M,1024] x W[1024,1024], W given pre-transposed Wt[n][k].
// grid = (M/32, 2), block = 256 (8 waves). Wave w owns a 32x64 tile at
// (row0, blockIdx.y*512 + w*64). All fragments are contiguous b128 global
// loads (A: 16-bit A-layout 8-half groups; B: same 8-half k groups as A).
__global__ void __launch_bounds__(256)
gemm_wmma(const _Float16* __restrict__ A, const _Float16* __restrict__ Wt,
          _Float16* __restrict__ Ch, float* __restrict__ Cf,
          const float* __restrict__ bias) {
  const int lane = threadIdx.x & 31;
  const int wave = threadIdx.x >> 5;
  const int ln = lane & 15;
  const int hi = lane >> 4;
  const int row0 = blockIdx.x * 32;
  const int n0 = blockIdx.y * 512 + wave * 64;

  v8f acc0[4], acc1[4];
#pragma unroll
  for (int nt = 0; nt < 4; ++nt) { acc0[nt] = zero8(); acc1[nt] = zero8(); }

  const _Float16* ap0 = A + (size_t)(row0 + ln) * DM;        // M-tile 0
  const _Float16* ap1 = A + (size_t)(row0 + 16 + ln) * DM;   // M-tile 1
  const _Float16* bp  = Wt + (size_t)(n0 + ln) * DM;         // N col base

  for (int k0 = 0; k0 < DM; k0 += 32) {
    v16h a0 = ld16(ap0 + k0 + hi * 8, ap0 + k0 + 16 + hi * 8);
    v16h a1 = ld16(ap1 + k0 + hi * 8, ap1 + k0 + 16 + hi * 8);
#pragma unroll
    for (int nt = 0; nt < 4; ++nt) {
      const _Float16* bb = bp + (size_t)nt * 16 * DM + k0 + hi * 8;
      v16h bf = ld16(bb, bb + 16);
      acc0[nt] = wmma16(a0, bf, acc0[nt]);
      acc1[nt] = wmma16(a1, bf, acc1[nt]);
    }
  }

#pragma unroll
  for (int nt = 0; nt < 4; ++nt) {
    const int ncol = n0 + nt * 16 + ln;
    const float bv = (bias != nullptr) ? bias[ncol] : 0.0f;
#pragma unroll
    for (int r = 0; r < 8; ++r) {
      const int m0 = row0 + r + 8 * hi;       // C layout: VGPR r <-> row r+8*hi
      const int m1 = m0 + 16;
      if (Cf) {
        Cf[(size_t)m0 * DM + ncol] = acc0[nt][r] + bv;
        Cf[(size_t)m1 * DM + ncol] = acc1[nt][r] + bv;
      } else {
        Ch[(size_t)m0 * DM + ncol] = (_Float16)acc0[nt][r];
        Ch[(size_t)m1 * DM + ncol] = (_Float16)acc1[nt][r];
      }
    }
  }
}

// ---------------------------- Flash attention ------------------------------
// One wave per (b, h, 16-row q tile). Scores computed transposed
// (S^T = K Q^T): lane holds column q = lane%16, VGPR r holds key r+8*hi, so
// S^T's C-layout coincides with the A-layout needed by the P*V wmma and P
// stays in registers. Online softmax runs in the exp2 domain.
// SCALE2 = (1/sqrt(64)) * log2(e)
#define SCALE2 0.18033688011112042f

template <bool MASKED>
static __device__ inline void attn_chunk(
    int j0, int i0, int iq, int ln, int hi,
    const _Float16* __restrict__ kbase, const _Float16* __restrict__ vbase,
    const v16h& qb0, const v16h& qb1,
    float& m_run, float& l_run, v8f& o0, v8f& o1, v8f& o2, v8f& o3) {
  // ----- S^T tiles: A = K rows (16 keys x 32 d), B = Q^T -----
  v8f s0 = zero8(), s1 = zero8();
  {
    const _Float16* kp = kbase + ((size_t)((j0 + ln) & (NDS - 1)) << 10);
    v16h ka0 = ld16(kp + hi * 8,      kp + 16 + hi * 8);
    v16h ka1 = ld16(kp + 32 + hi * 8, kp + 48 + hi * 8);
    s0 = wmma16(ka0, qb0, s0);
    s0 = wmma16(ka1, qb1, s0);
  }
  {
    const _Float16* kp = kbase + ((size_t)((j0 + 16 + ln) & (NDS - 1)) << 10);
    v16h ka0 = ld16(kp + hi * 8,      kp + 16 + hi * 8);
    v16h ka1 = ld16(kp + 32 + hi * 8, kp + 48 + hi * 8);
    s1 = wmma16(ka0, qb0, s1);
    s1 = wmma16(ka1, qb1, s1);
  }
  // ----- scale (+ band mask on boundary chunks), per-q running max -----
  float p0[8], p1[8];
  float mx = -1e30f;
#pragma unroll
  for (int r = 0; r < 8; ++r) {
    float v0 = s0[r] * SCALE2;
    float v1 = s1[r] * SCALE2;
    if (MASKED) {
      const int jj0 = j0 + r + 8 * hi;        // key of s0[r]
      if (jj0 < iq - 2)      v0 = -1e30f;
      if (jj0 + 16 < iq - 2) v1 = -1e30f;
    }
    p0[r] = v0; p1[r] = v1;
    mx = fmaxf(mx, fmaxf(v0, v1));
  }
  mx = fmaxf(mx, __shfl_xor(mx, 16, 32));     // combine key halves (same q)
  const float mn = fmaxf(m_run, mx);
  const float al = exp2f(m_run - mn);

  // ----- P = exp2(S2-m), packed directly into the P*V wmma A-fragment -----
  v16h pa;
  float rs = 0.0f;
#pragma unroll
  for (int r = 0; r < 8; ++r) {
    const float e0 = exp2f(p0[r] - mn);
    const float e1 = exp2f(p1[r] - mn);
    rs += e0 + e1;
    pa[r]     = (_Float16)e0;                 // kv = hi*8 + r      (tile 0)
    pa[r + 8] = (_Float16)e1;                 // kv = 16 + hi*8 + r (tile 1)
  }
  rs += __shfl_xor(rs, 16, 32);
  l_run = l_run * al + rs;
  m_run = mn;

  // ----- rescale O (rows q = r+8*hi need alpha of lane r+8*hi) -----
#pragma unroll
  for (int r = 0; r < 8; ++r) {
    const float aw = __shfl(al, r + 8 * hi, 32);
    o0[r] *= aw; o1[r] *= aw; o2[r] *= aw; o3[r] *= aw;
  }

  // ----- O += P * V via Vt (contiguous b128 loads per lane) -----
  const int kvb = (j0 + hi * 8) & (NDS - 1);
  const _Float16* vp = vbase + kvb;
  o0 = wmma16(pa, ld16(vp,              vp + 16),              o0);
  o1 = wmma16(pa, ld16(vp + (16 << 10), vp + (16 << 10) + 16), o1);
  o2 = wmma16(pa, ld16(vp + (32 << 10), vp + (32 << 10) + 16), o2);
  o3 = wmma16(pa, ld16(vp + (48 << 10), vp + (48 << 10) + 16), o3);
}

__global__ void __launch_bounds__(256)
attn_kernel(const _Float16* __restrict__ Q, const _Float16* __restrict__ K,
            const _Float16* __restrict__ Vt, _Float16* __restrict__ O) {
  const int lane = threadIdx.x & 31;
  const int wave = threadIdx.x >> 5;
  const int ln = lane & 15;
  const int hi = lane >> 4;

  const int wid = blockIdx.x * 8 + wave;  // 0..4095
  const int qt = wid & 127;               // QN/16
  const int h  = (wid >> 7) & 15;
  const int b  = wid >> 11;
  const int i0 = qt * 16;
  const int iq = i0 + ln;                 // this lane's query row

  // Q as B-fragments of S^T = K Q^T: col q = ln, d = hi*8 + 0..7, 16 + hi*8 + 0..7
  const _Float16* qp = Q + ((size_t)(b * QN + i0 + ln) << 10) + h * DH;
  const v16h qb0 = ld16(qp + hi * 8,      qp + 16 + hi * 8);       // d 0..31
  const v16h qb1 = ld16(qp + 32 + hi * 8, qp + 48 + hi * 8);  // d 32..63

  const _Float16* kbase = K + ((size_t)(b * NDS) << 10) + h * DH;
  const _Float16* vbase = Vt + ((size_t)((b * NH + h) * DH + ln) << 10);

  float m_run = -1e30f, l_run = 0.0f;
  v8f o0 = zero8(), o1 = zero8(), o2 = zero8(), o3 = zero8();

  int js = i0 - 2; if (js < 0) js = 0; js &= ~31;
  int jmid = js + 64; if (jmid > 2 * NDS) jmid = 2 * NDS;

  int j0 = js;
  for (; j0 < jmid; j0 += 32)       // boundary: band mask active
    attn_chunk<true>(j0, i0, iq, ln, hi, kbase, vbase, qb0, qb1,
                     m_run, l_run, o0, o1, o2, o3);
  for (; j0 < 2 * NDS; j0 += 32)    // steady state: fully allowed
    attn_chunk<false>(j0, i0, iq, ln, hi, kbase, vbase, qb0, qb1,
                      m_run, l_run, o0, o1, o2, o3);

  // ----- normalize + store O[b][q][h*64+d] as f16 -----
  _Float16* op = O + ((size_t)(b * QN + i0) << 10) + h * DH;
#pragma unroll
  for (int r = 0; r < 8; ++r) {
    const float inv = 1.0f / __shfl(l_run, r + 8 * hi, 32);
    const int q = r + 8 * hi;
    op[(size_t)q * DM + ln]      = (_Float16)(o0[r] * inv);
    op[(size_t)q * DM + 16 + ln] = (_Float16)(o1[r] * inv);
    op[(size_t)q * DM + 32 + ln] = (_Float16)(o2[r] * inv);
    op[(size_t)q * DM + 48 + ln] = (_Float16)(o3[r] * inv);
  }
}

// ---------------------------------------------------------------------------
extern "C" void kernel_launch(void* const* d_in, const int* in_sizes, int n_in,
                              void* d_out, int out_size, void* d_ws, size_t ws_size,
                              hipStream_t stream) {
  const float* x_full = (const float*)d_in[0];   // [2,2048,1024]
  const float* x_ds   = (const float*)d_in[1];   // [2,1024,1024]
  const float* wq     = (const float*)d_in[2];   // [1024,1024]
  const float* wk     = (const float*)d_in[3];
  const float* wv     = (const float*)d_in[4];
  const float* wo     = (const float*)d_in[5];
  const float* bo     = (const float*)d_in[6];   // [1024]
  float* out          = (float*)d_out;           // [2,2048,1024] f32

  // workspace layout (f16 halves); total ~48 MB
  _Float16* ws  = (_Float16*)d_ws;
  _Float16* Xf  = ws;                         // 4096*1024
  _Float16* Xd  = Xf  + (size_t)4096 * 1024;  // 2048*1024
  _Float16* Wqt = Xd  + (size_t)2048 * 1024;  // 1024*1024 (transposed [n][k])
  _Float16* Wkt = Wqt + (size_t)1024 * 1024;
  _Float16* Wvt = Wkt + (size_t)1024 * 1024;
  _Float16* Wot = Wvt + (size_t)1024 * 1024;
  _Float16* Qh  = Wot + (size_t)1024 * 1024;  // 4096*1024
  _Float16* Kh  = Qh  + (size_t)4096 * 1024;  // 2048*1024
  _Float16* Vh  = Kh  + (size_t)2048 * 1024;  // 2048*1024
  _Float16* Vth = Vh  + (size_t)2048 * 1024;  // 2048*1024 (B,H,64,NDS)
  _Float16* Oh  = Vth + (size_t)2048 * 1024;  // 4096*1024

  // 1) convert activations; convert+transpose weights
  cvt_f32_f16<<<4096, 256, 0, stream>>>(x_full, Xf, 4096 * 1024);
  cvt_f32_f16<<<2048, 256, 0, stream>>>(x_ds,   Xd, 2048 * 1024);
  cvt_transpose_w<<<4096, 256, 0, stream>>>(wq, Wqt);
  cvt_transpose_w<<<4096, 256, 0, stream>>>(wk, Wkt);
  cvt_transpose_w<<<4096, 256, 0, stream>>>(wv, Wvt);
  cvt_transpose_w<<<4096, 256, 0, stream>>>(wo, Wot);

  // 2) projections (WMMA GEMMs, LDS-free)
  gemm_wmma<<<dim3(128, 2), 256, 0, stream>>>(Xf, Wqt, Qh, nullptr, nullptr);
  gemm_wmma<<<dim3(64,  2), 256, 0, stream>>>(Xd, Wkt, Kh, nullptr, nullptr);
  gemm_wmma<<<dim3(64,  2), 256, 0, stream>>>(Xd, Wvt, Vh, nullptr, nullptr);

  // 3) Vt for contiguous PV B-fragments
  transpose_v<<<8192, 256, 0, stream>>>(Vh, Vth);

  // 4) banded flash attention (4096 waves, 8 per block, zero LDS)
  attn_kernel<<<512, 256, 0, stream>>>(Qh, Kh, Vth, Oh);

  // 5) output projection + bias -> f32 d_out
  gemm_wmma<<<dim3(128, 2), 256, 0, stream>>>(Oh, Wot, nullptr, out, bo);
}
